// MultiHeadAttention_31250182046090
// MI455X (gfx1250) — matrix operand order repaired and checked by reading, not compiled, not run on hardware
//
#include <hip/hip_runtime.h>
#include <hip/hip_bf16.h>

// ---------------------------------------------------------------------------
// MultiHeadAttention forward for gfx1250 (MI455X), bf16 WMMA pipeline.
//   d_out = [ output (B*SQ*E fp32) | attn_weights (B*H*SQ*SK fp32) ]
// Workspace (64 MB): Q16, K16 [b][h][s][d] bf16; V16T [b][h][d][s] bf16;
//                    Ctx16 [b][s][e] bf16.
// Fused attention kernel: per workgroup = one (b,h) x 16 query rows;
// scores -> softmax -> P*V all in LDS, attn weights written to HBM once.
// ---------------------------------------------------------------------------

#define B_   4
#define H_   16
#define SQ_  2048
#define SK_  2048
#define D_   64
#define E_   1024
#define MTOT 8192          // B_*SQ_
#define SPITCH 2056        // LDS strip pitch in floats (2048 + 8, bank skew)

typedef __attribute__((ext_vector_type(16))) __bf16 v16bf;
typedef __attribute__((ext_vector_type(8)))  __bf16 v8bf;
typedef __attribute__((ext_vector_type(8)))  float  v8f;
typedef __attribute__((ext_vector_type(4)))  float  v4f;

// ---- fragment loaders -----------------------------------------------------
// A-frag (16x32 bf16, row-major, leading dim ld):
// lane L: row = L&15; elems 0..7 = K[kb..kb+7], elems 8..15 = K[kb+16..kb+23],
// kb = (L<16) ? 0 : 8.   (ISA 7.12.2, 16-bit A 16x32)
__device__ __forceinline__ v16bf mha_load_a_bf16(const __bf16* base, int ld, int lane) {
  const __bf16* p0 = base + (size_t)(lane & 15) * ld + ((lane & 16) ? 8 : 0);
  v8bf x0 = *(const v8bf*)(p0);
  v8bf x1 = *(const v8bf*)(p0 + 16);
  return __builtin_shufflevector(x0, x1, 0,1,2,3,4,5,6,7,8,9,10,11,12,13,14,15);
}

__device__ __forceinline__ v16bf mha_load_a_f32(const float* base, int ld, int lane) {
  const float* p0 = base + (size_t)(lane & 15) * ld + ((lane & 16) ? 8 : 0);
  const float* p1 = p0 + 16;
  v4f a = *(const v4f*)(p0);
  v4f b = *(const v4f*)(p0 + 4);
  v4f c = *(const v4f*)(p1);
  v4f d = *(const v4f*)(p1 + 4);
  v16bf r;
#pragma unroll
  for (int i = 0; i < 4; ++i) {
    r[i]      = (__bf16)a[i];
    r[4 + i]  = (__bf16)b[i];
    r[8 + i]  = (__bf16)c[i];
    r[12 + i] = (__bf16)d[i];
  }
  return r;
}

// B-frag (32x16 bf16): source "column-contiguous": Bc[n][k] row-major.
// lane L: col = L&15; elems 0..7 = K[kb..kb+7], elems 8..15 = K[kb+16..kb+23], kb = (L<16) ? 0 : 8.
__device__ __forceinline__ v16bf mha_load_b_bf16(const __bf16* base, int ld, int lane) {
  const __bf16* p = base + (size_t)(lane & 15) * ld + ((lane & 16) ? 8 : 0);
  v8bf x0 = *(const v8bf*)(p);
  v8bf x1 = *(const v8bf*)(p + 16);
  return __builtin_shufflevector(x0, x1, 0,1,2,3,4,5,6,7,8,9,10,11,12,13,14,15);
}

__device__ __forceinline__ v16bf mha_load_b_f32(const float* base, int ld, int lane) {
  const float* p = base + (size_t)(lane & 15) * ld + ((lane & 16) ? 8 : 0);
  v4f a = *(const v4f*)(p);
  v4f b = *(const v4f*)(p + 4);
  v4f c = *(const v4f*)(p + 16);
  v4f d = *(const v4f*)(p + 20);
  v16bf r;
#pragma unroll
  for (int i = 0; i < 4; ++i) {
    r[i]      = (__bf16)a[i];
    r[4 + i]  = (__bf16)b[i];
    r[8 + i]  = (__bf16)c[i];
    r[12 + i] = (__bf16)d[i];
  }
  return r;
}

__device__ __forceinline__ v8f mha_wmma(v16bf a, v16bf b, v8f c) {
  return __builtin_amdgcn_wmma_f32_16x16x32_bf16(false, a, false, b, (short)0, c,
                                                 false, false);
}

// ---------------------------------------------------------------------------
// Projection: out = X(M x 1024) @ W^T + bias, stored bf16.
// MODE 0: headed layout  [b][h][s][d]   (for Q, K)
// MODE 1: headed transposed [b][h][d][s] (for V, so P*V B-frags are contiguous)
// block = 256 threads (8 waves, 2x4), block tile 64 x 128, wave tile 32 x 32.
// ---------------------------------------------------------------------------
template <int MODE>
__global__ void __launch_bounds__(256) mha_proj(const float* __restrict__ X,
                                                const float* __restrict__ W,
                                                const float* __restrict__ bias,
                                                __bf16* __restrict__ out) {
  const int lane = threadIdx.x & 31;
  const int wave = threadIdx.x >> 5;
  const int m0 = blockIdx.y * 64 + (wave >> 2) * 32;
  const int n0 = blockIdx.x * 128 + (wave & 3) * 32;

  v8f acc[2][2];
#pragma unroll
  for (int mi = 0; mi < 2; ++mi)
#pragma unroll
    for (int ni = 0; ni < 2; ++ni)
#pragma unroll
      for (int i = 0; i < 8; ++i) acc[mi][ni][i] = 0.0f;

  for (int k0 = 0; k0 < E_; k0 += 32) {
    v16bf a0 = mha_load_a_f32(X + (size_t)m0 * E_ + k0, E_, lane);
    v16bf a1 = mha_load_a_f32(X + (size_t)(m0 + 16) * E_ + k0, E_, lane);
    v16bf b0 = mha_load_b_f32(W + (size_t)n0 * E_ + k0, E_, lane);
    v16bf b1 = mha_load_b_f32(W + (size_t)(n0 + 16) * E_ + k0, E_, lane);
    acc[0][0] = mha_wmma(a0, b0, acc[0][0]);
    acc[0][1] = mha_wmma(a0, b1, acc[0][1]);
    acc[1][0] = mha_wmma(a1, b0, acc[1][0]);
    acc[1][1] = mha_wmma(a1, b1, acc[1][1]);
  }

#pragma unroll
  for (int mi = 0; mi < 2; ++mi) {
#pragma unroll
    for (int ni = 0; ni < 2; ++ni) {
      const int n = n0 + 16 * ni + (lane & 15);
      const float bn = bias[n];
      const int h = n >> 6, d = n & 63;
#pragma unroll
      for (int r = 0; r < 8; ++r) {
        const int m = m0 + 16 * mi + r + ((lane & 16) ? 8 : 0);
        const int bi = m >> 11, s = m & (SQ_ - 1);
        const float val = acc[mi][ni][r] + bn;
        if (MODE == 0)
          out[((size_t)(bi * H_ + h) * SQ_ + s) * D_ + d] = (__bf16)val;
        else
          out[((size_t)(bi * H_ + h) * D_ + d) * SK_ + s] = (__bf16)val;
      }
    }
  }
}

// ---------------------------------------------------------------------------
// Fused attention: scores + softmax + P*V for one (b,h) x 16 query rows.
// block = 256 threads (8 waves); wave w owns key-columns [w*256, w*256+256).
// LDS: strip [16][SPITCH] fp32 probs, acc [16][64] fp32 context partials,
//      redA/redB [256] row partials, inv [16].
// attn weights hit HBM exactly once (the mandatory fp32 output write).
// ---------------------------------------------------------------------------
#define SMEM_STRIP 0
#define SMEM_ACC   (16 * SPITCH)                 // 32896
#define SMEM_REDA  (SMEM_ACC + 16 * 64)          // +1024
#define SMEM_REDB  (SMEM_REDA + 256)
#define SMEM_INV   (SMEM_REDB + 256)
#define SMEM_FLOATS (SMEM_INV + 16)
#define FUSED_SMEM_BYTES (SMEM_FLOATS * 4)       // 137,792 bytes

__global__ void __launch_bounds__(256) mha_attn_fused(const __bf16* __restrict__ Q,
                                                      const __bf16* __restrict__ K,
                                                      const __bf16* __restrict__ VT,
                                                      float* __restrict__ attn,
                                                      __bf16* __restrict__ ctx) {
  extern __shared__ float smem[];
  float* sP   = smem + SMEM_STRIP;   // [16][SPITCH]
  float* sAcc = smem + SMEM_ACC;     // [16][64]
  float* sRedA = smem + SMEM_REDA;   // [16][16]
  float* sRedB = smem + SMEM_REDB;   // [16][16]
  float* sInv  = smem + SMEM_INV;    // [16]

  const int t    = threadIdx.x;
  const int lane = t & 31;
  const int wave = t >> 5;             // 0..7
  const int z    = blockIdx.y;         // b*H + h
  const int m0   = blockIdx.x * 16;    // query-row base
  const int bi   = z >> 4, h = z & 15;

  const __bf16* Qb = Q + (size_t)z * SQ_ * D_;
  const __bf16* Kb = K + (size_t)z * SK_ * D_;
  const __bf16* Vb = VT + (size_t)z * D_ * SK_;
  float* Sb = attn + (size_t)z * SQ_ * SK_;

  // zero the context accumulator while scores are being computed
  for (int i = t; i < 16 * 64; i += 256) sAcc[i] = 0.0f;

  // ---- phase 1: scores strip (16 x 2048), wave w -> keys [w*256, w*256+256)
  {
    v8f acc[16];
#pragma unroll
    for (int ni = 0; ni < 16; ++ni)
#pragma unroll
      for (int i = 0; i < 8; ++i) acc[ni][i] = 0.0f;

#pragma unroll
    for (int k0 = 0; k0 < D_; k0 += 32) {
      v16bf a = mha_load_a_bf16(Qb + (size_t)m0 * D_ + k0, D_, lane);
#pragma unroll
      for (int ni = 0; ni < 16; ++ni) {
        v16bf b = mha_load_b_bf16(Kb + (size_t)(wave * 256 + ni * 16) * D_ + k0,
                                  D_, lane);
        acc[ni] = mha_wmma(a, b, acc[ni]);
      }
    }

    const float scale = 0.125f;  // HEAD_DIM^-0.5
#pragma unroll
    for (int ni = 0; ni < 16; ++ni) {
      const int col = wave * 256 + ni * 16 + (lane & 15);
#pragma unroll
      for (int r = 0; r < 8; ++r) {
        const int row = r + ((lane & 16) ? 8 : 0);
        sP[row * SPITCH + col] = acc[ni][r] * scale;
      }
    }
  }
  __syncthreads();

  // ---- phase 2: softmax over each of the 16 rows (2048 wide).
  // thread t: row = t>>4, chunk = t&15 -> 128 consecutive columns.
  const int row   = t >> 4;
  const int chunk = t & 15;
  float* prow = sP + row * SPITCH + chunk * 128;
  {
    float pm = -3.402823466e38f;
    for (int j = 0; j < 128; ++j) pm = fmaxf(pm, prow[j]);
    sRedA[t] = pm;
  }
  __syncthreads();
  {
    float rowmax = -3.402823466e38f;
#pragma unroll
    for (int i = 0; i < 16; ++i) rowmax = fmaxf(rowmax, sRedA[(t & ~15) + i]);
    float s = 0.0f;
    for (int j = 0; j < 128; ++j) {
      const float e = __expf(prow[j] - rowmax);
      prow[j] = e;
      s += e;
    }
    sRedB[t] = s;
  }
  __syncthreads();
  if (t < 16) {
    float s = 0.0f;
#pragma unroll
    for (int i = 0; i < 16; ++i) s += sRedB[t * 16 + i];
    sInv[t] = 1.0f / s;
  }
  __syncthreads();

  // ---- phase 2d: normalize strip in place + single coalesced HBM write of
  // the attn weights (float4 per thread per iteration).
  for (int i = 4 * t; i < 16 * 2048; i += 1024) {
    const int r = i >> 11, c = i & 2047;
    float* ps = sP + r * SPITCH + c;
    v4f val = *(v4f*)ps;
    val *= sInv[r];
    *(v4f*)ps = val;
    *(v4f*)(Sb + (size_t)(m0 + r) * SK_ + c) = val;
  }
  __syncthreads();

  // ---- phase 3: context = P(16x2048) @ V(2048x64); wave w does k-slice
  // [w*256, w*256+256), then reduces via LDS float atomics.
  {
    v8f cacc[4];
#pragma unroll
    for (int ni = 0; ni < 4; ++ni)
#pragma unroll
      for (int i = 0; i < 8; ++i) cacc[ni][i] = 0.0f;

#pragma unroll
    for (int kk = 0; kk < 8; ++kk) {
      const int k0 = wave * 256 + kk * 32;
      v16bf a = mha_load_a_f32(sP + k0, SPITCH, lane);
#pragma unroll
      for (int ni = 0; ni < 4; ++ni) {
        v16bf b = mha_load_b_bf16(Vb + (size_t)(ni * 16) * SK_ + k0, SK_, lane);
        cacc[ni] = mha_wmma(a, b, cacc[ni]);
      }
    }

#pragma unroll
    for (int ni = 0; ni < 4; ++ni) {
      const int d = ni * 16 + (lane & 15);
#pragma unroll
      for (int r = 0; r < 8; ++r) {
        const int rr = r + ((lane & 16) ? 8 : 0);
        atomicAdd(&sAcc[rr * 64 + d], cacc[ni][r]);
      }
    }
  }
  __syncthreads();

  // ---- write bf16 context, token-major [b][s][e], e = h*64 + d
  for (int i = t; i < 16 * 64; i += 256) {
    const int rr = i >> 6, d = i & 63;
    ctx[((size_t)bi * SQ_ + (m0 + rr)) * E_ + h * 64 + d] = (__bf16)sAcc[i];
  }
}

// ---------------------------------------------------------------------------
// Output projection: out = Ctx (M x 1024 bf16) @ Wo^T + bo, fp32 out.
// ---------------------------------------------------------------------------
__global__ void __launch_bounds__(256) mha_outproj(const __bf16* __restrict__ C,
                                                   const float* __restrict__ W,
                                                   const float* __restrict__ bias,
                                                   float* __restrict__ out) {
  const int lane = threadIdx.x & 31;
  const int wave = threadIdx.x >> 5;
  const int m0 = blockIdx.y * 64 + (wave >> 2) * 32;
  const int n0 = blockIdx.x * 128 + (wave & 3) * 32;

  v8f acc[2][2];
#pragma unroll
  for (int mi = 0; mi < 2; ++mi)
#pragma unroll
    for (int ni = 0; ni < 2; ++ni)
#pragma unroll
      for (int i = 0; i < 8; ++i) acc[mi][ni][i] = 0.0f;

  for (int k0 = 0; k0 < E_; k0 += 32) {
    v16bf a0 = mha_load_a_bf16(C + (size_t)m0 * E_ + k0, E_, lane);
    v16bf a1 = mha_load_a_bf16(C + (size_t)(m0 + 16) * E_ + k0, E_, lane);
    v16bf b0 = mha_load_b_f32(W + (size_t)n0 * E_ + k0, E_, lane);
    v16bf b1 = mha_load_b_f32(W + (size_t)(n0 + 16) * E_ + k0, E_, lane);
    acc[0][0] = mha_wmma(a0, b0, acc[0][0]);
    acc[0][1] = mha_wmma(a0, b1, acc[0][1]);
    acc[1][0] = mha_wmma(a1, b0, acc[1][0]);
    acc[1][1] = mha_wmma(a1, b1, acc[1][1]);
  }

#pragma unroll
  for (int mi = 0; mi < 2; ++mi) {
#pragma unroll
    for (int ni = 0; ni < 2; ++ni) {
      const int n = n0 + 16 * ni + (lane & 15);
      const float bn = bias[n];
#pragma unroll
      for (int r = 0; r < 8; ++r) {
        const int m = m0 + 16 * mi + r + ((lane & 16) ? 8 : 0);
        out[(size_t)m * E_ + n] = acc[mi][ni][r] + bn;
      }
    }
  }
}

// ---------------------------------------------------------------------------
extern "C" void kernel_launch(void* const* d_in, const int* in_sizes, int n_in,
                              void* d_out, int out_size, void* d_ws, size_t ws_size,
                              hipStream_t stream) {
  const float* q  = (const float*)d_in[0];
  const float* k  = (const float*)d_in[1];
  const float* v  = (const float*)d_in[2];
  const float* Wq = (const float*)d_in[3];
  const float* bq = (const float*)d_in[4];
  const float* Wk = (const float*)d_in[5];
  const float* bk = (const float*)d_in[6];
  const float* Wv = (const float*)d_in[7];
  const float* bv = (const float*)d_in[8];
  const float* Wo = (const float*)d_in[9];
  const float* bo = (const float*)d_in[10];

  float* out  = (float*)d_out;
  float* attn = out + (size_t)B_ * SQ_ * E_;  // outputs then attn weights

  const size_t HEADED = (size_t)B_ * H_ * SQ_ * D_;  // 8,388,608 elems
  __bf16* Q16  = (__bf16*)d_ws;
  __bf16* K16  = Q16 + HEADED;
  __bf16* V16T = K16 + HEADED;
  __bf16* C16  = V16T + HEADED;  // total 64 MB of bf16 workspace

  const dim3 projGrid(E_ / 128, MTOT / 64);   // (8, 128)
  const dim3 fusedGrid(SQ_ / 16, B_ * H_);    // (128, 64) = 8192 WGs

  mha_proj<0><<<projGrid, 256, 0, stream>>>(q, Wq, bq, Q16);
  mha_proj<0><<<projGrid, 256, 0, stream>>>(k, Wk, bk, K16);
  mha_proj<1><<<projGrid, 256, 0, stream>>>(v, Wv, bv, V16T);
  mha_attn_fused<<<fusedGrid, 256, FUSED_SMEM_BYTES, stream>>>(Q16, K16, V16T,
                                                               attn, C16);
  mha_outproj<<<projGrid, 256, 0, stream>>>(C16, Wo, bo, out);
}
